// scGAT_customforward_40106404610261
// MI455X (gfx1250) — matrix operand order repaired and checked by reading, not compiled, not run on hardware
//
#include <hip/hip_runtime.h>

// ---------------- constants from the reference ----------------
#define NN      50000
#define EE      800000
#define ETOT    (EE + NN)       // self loops appended
#define HEADS   8
#define HID     8
#define DIMIN   1000
#define DIMOUT  32
#define NEG_SLOPE 0.2f

// ---------------- WMMA types ----------------
typedef __attribute__((ext_vector_type(16))) __bf16 v16bf;
typedef __attribute__((ext_vector_type(8)))  float  v8f;
typedef __attribute__((ext_vector_type(4)))  int    v4i;

union Frag16 { unsigned int u[8]; v16bf v; };

#if __has_builtin(__builtin_amdgcn_global_load_async_to_lds_b128) && \
    __has_builtin(__builtin_amdgcn_s_wait_asynccnt)
#define HAVE_ASYNC_LDS 1
typedef __attribute__((address_space(1))) v4i gv4i;   // global int4
typedef __attribute__((address_space(3))) v4i lv4i;   // LDS int4
#endif

__device__ __forceinline__ unsigned short f2bf(float f) {
    unsigned int b = __float_as_uint(f);
    unsigned int r = (b + 0x7FFFu + ((b >> 16) & 1u)) >> 16;   // RNE
    return (unsigned short)r;
}
__device__ __forceinline__ unsigned int pack2(float lo, float hi) {
    return (unsigned int)f2bf(lo) | ((unsigned int)f2bf(hi) << 16);
}

// guarded load of 8 consecutive K elements of row `row` from split A = [A1|A2],
// zero-padded past Ktot (boundary/tail chunks only)
__device__ __forceinline__ void loadRun8(const float* __restrict__ A1, int K1,
                                         const float* __restrict__ A2, int K2,
                                         int Ktot, long row, int k0,
                                         unsigned int* out4) {
    float f[8];
    #pragma unroll
    for (int i = 0; i < 8; ++i) {
        int k = k0 + i;
        float v = 0.0f;
        if (k < Ktot)
            v = (k < K1) ? A1[row * (long)K1 + k] : A2[row * (long)K2 + (k - K1)];
        f[i] = v;
    }
    #pragma unroll
    for (int i = 0; i < 4; ++i) out4[i] = pack2(f[2*i], f[2*i+1]);
}

// ---- one-time weight transform: WT[n][k] = bf16(W[k][n]), zero-padded to KP ----
__global__ void convert_WT(const float* __restrict__ W, int Ntot, int Ktot, int KP,
                           unsigned short* __restrict__ WT) {
    long idx = (long)blockIdx.x * blockDim.x + threadIdx.x;
    if (idx >= (long)Ntot * KP) return;
    int n = (int)(idx / KP);
    int k = (int)(idx % KP);
    WT[idx] = (k < Ktot) ? f2bf(W[(long)k * Ntot + n]) : (unsigned short)0;
}

// ---------------- bf16 WMMA GEMM: Out[M, Ntot] = [A1 | A2] @ W ----------------
// 256 threads = 8 waves; wave w -> M-tile blockIdx.x*8+w; each wave does 4 N-tiles
// (columns [blockIdx.y*64, +64)). B chunks staged into double-buffered LDS from the
// pre-transposed zero-padded WT via async global->LDS b128 copies, pipelined one
// chunk ahead of the WMMA compute (s_wait_asynccnt at loop bottom).
__global__ __launch_bounds__(256)
void gemm_wmma_bf16(const float* __restrict__ A1, int K1,
                    const float* __restrict__ A2, int K2,
                    const unsigned short* __restrict__ WT, int KP,
                    int Ntot, float* __restrict__ Out, int Mrows) {
    const int Ktot = K1 + K2;
    const int tid  = threadIdx.x;
    const int wave = tid >> 5;
    const int lane = tid & 31;
    const int mtile = blockIdx.x * 8 + wave;
    const int colbase = blockIdx.y * 64;
    const int Mtiles = Mrows >> 4;                       // Mrows % 16 == 0 here
    const bool mOK = __builtin_amdgcn_readfirstlane(mtile) < Mtiles; // scalar guard

    __shared__ __align__(16) unsigned int ldsB[2][64 * 16];  // [buf][n][k] bf16 pairs

    // B-staging assignment: lane copies 16B = 8 bf16 of column nl, k-slice part*8
    const int nl   = tid >> 2;      // 0..63
    const int part = tid & 3;       // 0..3

    const int lrow  = lane & 15;
    const int lhalf = lane >> 4;    // 0 or 1
    const long row  = (long)mtile * 16 + lrow;

    auto stageB = [&](int ch, int buf) {
        const unsigned short* gsrc = WT + (long)(colbase + nl) * KP + (ch << 5) + part * 8;
        unsigned short*       ldst = (unsigned short*)ldsB[buf] + nl * 32 + part * 8;
#ifdef HAVE_ASYNC_LDS
        __builtin_amdgcn_global_load_async_to_lds_b128((gv4i*)gsrc, (lv4i*)ldst, 0, 0);
#else
        *(uint4*)ldst = *(const uint4*)gsrc;
#endif
    };

    v8f acc0 = {}, acc1 = {}, acc2 = {}, acc3 = {};

    const int nchunks = KP >> 5;    // WT zero-padded -> no B tail guards
    stageB(0, 0);
#ifdef HAVE_ASYNC_LDS
    __builtin_amdgcn_s_wait_asynccnt(0);
#endif
    __syncthreads();

    for (int ch = 0; ch < nchunks; ++ch) {
        const int kc  = ch << 5;
        const int cur = ch & 1;
        if (ch + 1 < nchunks) stageB(ch + 1, cur ^ 1);   // prefetch next B chunk

        // ---- A fragment (scalar-uniform fast/guarded selection) ----
        Frag16 afr;
        const bool fastA = (kc + 32 <= K1) || (kc >= K1 && kc + 32 <= Ktot);
        if (mOK && fastA) {
            const float* bp = (kc < K1) ? (A1 + row * (long)K1 + kc)
                                        : (A2 + row * (long)K2 + (kc - K1));
            const float4* p = (const float4*)bp;
            float4 a = p[lhalf * 4], b = p[lhalf * 4 + 1];
            float4 c = p[lhalf * 4 + 2], d = p[lhalf * 4 + 3];
            afr.u[0] = pack2(a.x, a.y); afr.u[1] = pack2(a.z, a.w);
            afr.u[2] = pack2(b.x, b.y); afr.u[3] = pack2(b.z, b.w);
            afr.u[4] = pack2(c.x, c.y); afr.u[5] = pack2(c.z, c.w);
            afr.u[6] = pack2(d.x, d.y); afr.u[7] = pack2(d.z, d.w);
        } else if (mOK) {
            loadRun8(A1, K1, A2, K2, Ktot, row, kc + lhalf * 16,     &afr.u[0]);
            loadRun8(A1, K1, A2, K2, Ktot, row, kc + lhalf * 16 + 8, &afr.u[4]);
        } else {
            #pragma unroll
            for (int i = 0; i < 8; ++i) afr.u[i] = 0u;
        }

        // ---- B fragments from LDS + 4 WMMAs ----
        const unsigned int* lb = ldsB[cur];
        {
            Frag16 bf;
            int bb = (0 * 16 + lrow) * 16 + lhalf * 8;
            #pragma unroll
            for (int v = 0; v < 8; ++v) bf.u[v] = lb[bb + v];
            acc0 = __builtin_amdgcn_wmma_f32_16x16x32_bf16(false, afr.v, false, bf.v,
                                                           (short)0, acc0, false, false);
        }
        {
            Frag16 bf;
            int bb = (1 * 16 + lrow) * 16 + lhalf * 8;
            #pragma unroll
            for (int v = 0; v < 8; ++v) bf.u[v] = lb[bb + v];
            acc1 = __builtin_amdgcn_wmma_f32_16x16x32_bf16(false, afr.v, false, bf.v,
                                                           (short)0, acc1, false, false);
        }
        {
            Frag16 bf;
            int bb = (2 * 16 + lrow) * 16 + lhalf * 8;
            #pragma unroll
            for (int v = 0; v < 8; ++v) bf.u[v] = lb[bb + v];
            acc2 = __builtin_amdgcn_wmma_f32_16x16x32_bf16(false, afr.v, false, bf.v,
                                                           (short)0, acc2, false, false);
        }
        {
            Frag16 bf;
            int bb = (3 * 16 + lrow) * 16 + lhalf * 8;
            #pragma unroll
            for (int v = 0; v < 8; ++v) bf.u[v] = lb[bb + v];
            acc3 = __builtin_amdgcn_wmma_f32_16x16x32_bf16(false, afr.v, false, bf.v,
                                                           (short)0, acc3, false, false);
        }

#ifdef HAVE_ASYNC_LDS
        __builtin_amdgcn_s_wait_asynccnt(0);   // next-chunk copy landed
#endif
        __syncthreads();                        // all waves done reading/writing
    }

    // store: element r of acc -> (M = mtile*16 + r + 8*lhalf, N = colbase + nt*16 + lrow)
    if (mOK) {
        #pragma unroll
        for (int r = 0; r < 8; ++r) {
            long m = (long)mtile * 16 + r + 8 * lhalf;
            Out[m * Ntot + colbase +  0 + lrow] = acc0[r];
            Out[m * Ntot + colbase + 16 + lrow] = acc1[r];
            Out[m * Ntot + colbase + 32 + lrow] = acc2[r];
            Out[m * Ntot + colbase + 48 + lrow] = acc3[r];
        }
    }
}

// ---------------- attention dots: als/ald[n,h] = sum_c h[n,h,c] * a[h,c] ----------------
__global__ void attn_dots(const float* __restrict__ h, const float* __restrict__ as_,
                          const float* __restrict__ ad_, float* __restrict__ als,
                          float* __restrict__ ald, int Nn, int C) {
    int idx = blockIdx.x * blockDim.x + threadIdx.x;   // n*H + hd
    if (idx >= Nn * HEADS) return;
    int n = idx >> 3, hd = idx & 7;
    const float* hp  = h   + (long)n * HEADS * C + hd * C;
    const float* asp = as_ + hd * C;
    const float* adp = ad_ + hd * C;
    float s = 0.0f, d = 0.0f;
    for (int c = 0; c < C; ++c) { float v = hp[c]; s += v * asp[c]; d += v * adp[c]; }
    als[idx] = s; ald[idx] = d;
}

__global__ void fill_f32(float* __restrict__ p, long n, float v) {
    long i = (long)blockIdx.x * blockDim.x + threadIdx.x;
    if (i < n) p[i] = v;
}

__device__ __forceinline__ void getEdge(const int* __restrict__ ei, int e, int& s, int& d) {
    if (e < EE) { s = ei[e]; d = ei[EE + e]; }
    else        { s = e - EE; d = e - EE; }
}

__device__ __forceinline__ float lrelu(float v) { return v > 0.0f ? v : NEG_SLOPE * v; }

__device__ __forceinline__ void atomicMaxF(float* addr, float v) {
    if (v >= 0.0f) atomicMax((int*)addr, __float_as_int(v));
    else           atomicMin((unsigned int*)addr, __float_as_uint(v));
}

__global__ void edge_max(const int* __restrict__ ei, const float* __restrict__ als,
                         const float* __restrict__ ald, float* __restrict__ maxb) {
    int e = blockIdx.x * blockDim.x + threadIdx.x;
    if (e >= ETOT) return;
    int s, d; getEdge(ei, e, s, d);
    const float4* ps = (const float4*)(als + (long)s * HEADS);
    const float4* pd = (const float4*)(ald + (long)d * HEADS);
    float4 s0 = ps[0], s1 = ps[1], d0 = pd[0], d1 = pd[1];
    float v[HEADS] = { s0.x+d0.x, s0.y+d0.y, s0.z+d0.z, s0.w+d0.w,
                       s1.x+d1.x, s1.y+d1.y, s1.z+d1.z, s1.w+d1.w };
    #pragma unroll
    for (int h = 0; h < HEADS; ++h) atomicMaxF(&maxb[(long)d * HEADS + h], lrelu(v[h]));
}

__global__ void edge_expsum(const int* __restrict__ ei, const float* __restrict__ als,
                            const float* __restrict__ ald, const float* __restrict__ maxb,
                            float* __restrict__ eebuf, float* __restrict__ denom) {
    int e = blockIdx.x * blockDim.x + threadIdx.x;
    if (e >= ETOT) return;
    int s, d; getEdge(ei, e, s, d);
    const float4* ps = (const float4*)(als + (long)s * HEADS);
    const float4* pd = (const float4*)(ald + (long)d * HEADS);
    float4 s0 = ps[0], s1 = ps[1], d0 = pd[0], d1 = pd[1];
    float v[HEADS] = { s0.x+d0.x, s0.y+d0.y, s0.z+d0.z, s0.w+d0.w,
                       s1.x+d1.x, s1.y+d1.y, s1.z+d1.z, s1.w+d1.w };
    #pragma unroll
    for (int h = 0; h < HEADS; ++h) {
        float ex = __expf(lrelu(v[h]) - maxb[(long)d * HEADS + h]);
        eebuf[(long)e * HEADS + h] = ex;
        atomicAdd(&denom[(long)d * HEADS + h], ex);
    }
}

__global__ void edge_aggr(const int* __restrict__ ei, const float* __restrict__ hsrc,
                          const float* __restrict__ eebuf, const float* __restrict__ denom,
                          float* __restrict__ acc, int C) {
    long idx = (long)blockIdx.x * blockDim.x + threadIdx.x;   // e*H + h
    if (idx >= (long)ETOT * HEADS) return;
    int e = (int)(idx >> 3), h = (int)(idx & 7);
    int s, d; getEdge(ei, e, s, d);
    float alpha = eebuf[idx] / (denom[(long)d * HEADS + h] + 1e-16f);
    const float* hp = hsrc + (long)s * HEADS * C + h * C;
    float*       ap = acc  + (long)d * HEADS * C + h * C;
    for (int c = 0; c < C; ++c) atomicAdd(&ap[c], hp[c] * alpha);
}

__global__ void bias_elu64(float* __restrict__ acc, const float* __restrict__ b, int Nn) {
    long idx = (long)blockIdx.x * blockDim.x + threadIdx.x;
    if (idx >= (long)Nn * 64) return;
    float v = acc[idx] + b[idx & 63];
    acc[idx] = v > 0.0f ? v : (__expf(v) - 1.0f);
}

// mean over heads + bias + log_softmax over 32 cols; one wave32 per node
__global__ __launch_bounds__(128)
void finalize_out(const float* __restrict__ acc3, const float* __restrict__ b3,
                  float* __restrict__ out, int Nn) {
    int node = blockIdx.x * 4 + (threadIdx.x >> 5);
    int j = threadIdx.x & 31;
    if (node >= Nn) return;
    float v = 0.0f;
    #pragma unroll
    for (int h = 0; h < HEADS; ++h) v += acc3[(long)node * 256 + h * 32 + j];
    v = v * 0.125f + b3[j];
    float m = v;
    #pragma unroll
    for (int off = 16; off; off >>= 1) m = fmaxf(m, __shfl_xor(m, off, 32));
    float ex = __expf(v - m);
    float ssum = ex;
    #pragma unroll
    for (int off = 16; off; off >>= 1) ssum += __shfl_xor(ssum, off, 32);
    out[(long)node * 32 + j] = v - m - __logf(ssum);
}

// ---------------- driver ----------------
extern "C" void kernel_launch(void* const* d_in, const int* in_sizes, int n_in,
                              void* d_out, int out_size, void* d_ws, size_t ws_size,
                              hipStream_t stream) {
    (void)in_sizes; (void)n_in; (void)out_size; (void)ws_size;
    const float* x   = (const float*)d_in[0];
    const int*   ei  = (const int*)  d_in[1];
    const float* W1  = (const float*)d_in[2];
    const float* as1 = (const float*)d_in[3];
    const float* ad1 = (const float*)d_in[4];
    const float* b1  = (const float*)d_in[5];
    const float* W2  = (const float*)d_in[6];
    const float* as2 = (const float*)d_in[7];
    const float* ad2 = (const float*)d_in[8];
    const float* b2  = (const float*)d_in[9];
    const float* W3  = (const float*)d_in[10];
    const float* as3 = (const float*)d_in[11];
    const float* ad3 = (const float*)d_in[12];
    const float* b3  = (const float*)d_in[13];
    float* out = (float*)d_out;

    // workspace layout (floats, then bf16 weight copies)
    float* ws = (float*)d_ws;
    float* hA    = ws;                          // N*64  (GEMM h, layers 1&2)
    float* accA  = hA   + (long)NN * 64;        // N*64  (aggregate / activation, layers 1&2)
    float* h3    = accA + (long)NN * 64;        // N*256 (layer-3 GEMM h)
    float* acc3  = h3   + (long)NN * 256;       // N*256 (layer-3 aggregate)
    float* als   = acc3 + (long)NN * 256;       // N*8
    float* ald   = als  + (long)NN * HEADS;     // N*8
    float* maxb  = ald  + (long)NN * HEADS;     // N*8
    float* denom = maxb + (long)NN * HEADS;     // N*8
    float* eebuf = denom+ (long)NN * HEADS;     // ETOT*8
    unsigned short* WT1 = (unsigned short*)(eebuf + (long)ETOT * HEADS); // 64*1024
    unsigned short* WT2 = WT1 + 64 * 1024;                               // 64*1088
    unsigned short* WT3 = WT2 + 64 * 1088;                               // 256*64

    const int KP1 = 1024, KP2 = 1088, KP3 = 64;

    const int MT = NN / 16;                     // 3125 M-tiles
    dim3 gemmBlk(256);
    dim3 gemmGrd64((MT + 7) / 8, 1);
    dim3 gemmGrd256((MT + 7) / 8, 4);
    const int T = 256;
    const int nhBlocks   = (NN * HEADS + T - 1) / T;
    const int edgeBlocks = (ETOT + T - 1) / T;
    const long ehN = (long)ETOT * HEADS;
    const int ehBlocks = (int)((ehN + T - 1) / T);
    const float NEGBIG = -3.4028235e38f;

    auto fill = [&](float* p, long n, float v) {
        fill_f32<<<(int)((n + T - 1) / T), T, 0, stream>>>(p, n, v);
    };

    // one-time weight transforms (transposed, zero-padded bf16)
    convert_WT<<<(64 * KP1 + T - 1) / T, T, 0, stream>>>(W1, 64,  DIMIN,     KP1, WT1);
    convert_WT<<<(64 * KP2 + T - 1) / T, T, 0, stream>>>(W2, 64,  DIMIN+64,  KP2, WT2);
    convert_WT<<<(256 * KP3 + T - 1) / T, T, 0, stream>>>(W3, 256, 64,       KP3, WT3);

    // ---------------- Layer 1 ----------------
    gemm_wmma_bf16<<<gemmGrd64, gemmBlk, 0, stream>>>(x, DIMIN, nullptr, 0, WT1, KP1, 64, hA, NN);
    attn_dots<<<nhBlocks, T, 0, stream>>>(hA, as1, ad1, als, ald, NN, HID);
    fill(maxb,  (long)NN * HEADS, NEGBIG);
    fill(denom, (long)NN * HEADS, 0.0f);
    fill(accA,  (long)NN * 64,    0.0f);
    edge_max   <<<edgeBlocks, T, 0, stream>>>(ei, als, ald, maxb);
    edge_expsum<<<edgeBlocks, T, 0, stream>>>(ei, als, ald, maxb, eebuf, denom);
    edge_aggr  <<<ehBlocks,  T, 0, stream>>>(ei, hA, eebuf, denom, accA, HID);
    bias_elu64 <<<(int)(((long)NN * 64 + T - 1) / T), T, 0, stream>>>(accA, b1, NN);

    // ---------------- Layer 2 (input = [x | h1]) ----------------
    gemm_wmma_bf16<<<gemmGrd64, gemmBlk, 0, stream>>>(x, DIMIN, accA, 64, WT2, KP2, 64, hA, NN);
    attn_dots<<<nhBlocks, T, 0, stream>>>(hA, as2, ad2, als, ald, NN, HID);
    fill(maxb,  (long)NN * HEADS, NEGBIG);
    fill(denom, (long)NN * HEADS, 0.0f);
    fill(accA,  (long)NN * 64,    0.0f);          // h1 dead after GEMM2 -> reuse
    edge_max   <<<edgeBlocks, T, 0, stream>>>(ei, als, ald, maxb);
    edge_expsum<<<edgeBlocks, T, 0, stream>>>(ei, als, ald, maxb, eebuf, denom);
    edge_aggr  <<<ehBlocks,  T, 0, stream>>>(ei, hA, eebuf, denom, accA, HID);
    bias_elu64 <<<(int)(((long)NN * 64 + T - 1) / T), T, 0, stream>>>(accA, b2, NN);

    // ---------------- Layer 3 (mean heads + log_softmax) ----------------
    gemm_wmma_bf16<<<gemmGrd256, gemmBlk, 0, stream>>>(accA, 64, nullptr, 0, WT3, KP3, 256, h3, NN);
    attn_dots<<<nhBlocks, T, 0, stream>>>(h3, as3, ad3, als, ald, NN, DIMOUT);
    fill(maxb,  (long)NN * HEADS, NEGBIG);
    fill(denom, (long)NN * HEADS, 0.0f);
    fill(acc3,  (long)NN * 256,   0.0f);
    edge_max   <<<edgeBlocks, T, 0, stream>>>(ei, als, ald, maxb);
    edge_expsum<<<edgeBlocks, T, 0, stream>>>(ei, als, ald, maxb, eebuf, denom);
    edge_aggr  <<<ehBlocks,  T, 0, stream>>>(ei, h3, eebuf, denom, acc3, DIMOUT);
    finalize_out<<<(NN + 3) / 4, 128, 0, stream>>>(acc3, b3, out, NN);
}
